// MHA_64338610094404
// MI455X (gfx1250) — matrix operand order repaired and checked by reading, not compiled, not run on hardware
//
#include <hip/hip_runtime.h>

typedef __attribute__((ext_vector_type(16))) _Float16 v16h;
typedef __attribute__((ext_vector_type(8)))  _Float16 h8;
typedef __attribute__((ext_vector_type(8)))  float    v8f;

#define LOG2E 1.4426950408889634f
// ds_swizzle imm: offset[15]=0 -> group-of-32: and=0x1f, or=0, xor=imm[14:10]
#define SWZ_XOR(v, x) __int_as_float(__builtin_amdgcn_ds_swizzle(__float_as_int(v), ((x) << 10) | 0x1f))

constexpr int Bc = 2, Hc = 16, Sc = 2048, Dk = 64;
constexpr int QTILE = 64;   // 4 waves x 16 q-rows per block
constexpr int KTILE = 64;   // keys staged per iteration (4 x 16-key subtiles)

// padded LDS row strides (multiples of 8 halves = 16B; odd multiples of 4 banks -> conflict-free)
constexpr int KS = 72;  // Ksh row stride (halves): 144 B
constexpr int VS = 72;  // Vt  row stride (halves): 144 B
constexpr int PS = 72;  // Psh row stride (halves): 144 B

static __device__ __forceinline__ v16h cat16(h8 lo, h8 hi) {
  union { h8 h[2]; v16h v; } u;
  u.h[0] = lo; u.h[1] = hi;
  return u.v;
}

static __device__ __forceinline__ v8f wmma_f16(v16h a, v16h b, v8f c) {
  // D = A(16x32 f16) * B(32x16 f16) + C(16x16 f32)
  return __builtin_amdgcn_wmma_f32_16x16x32_f16(false, a, false, b, (short)0, c,
                                                false, false);
}

__global__ __launch_bounds__(128) void fa_fwd_kernel(const float* __restrict__ Qg,
                                                     const float* __restrict__ Kg,
                                                     const float* __restrict__ Vg,
                                                     float* __restrict__ Og) {
  __shared__ alignas(16) _Float16 Ksh[KTILE][KS];     // key-major, dk contiguous
  __shared__ alignas(16) _Float16 Vt[Dk][VS];         // dk-major, key contiguous (transposed)
  __shared__ alignas(16) _Float16 Psh[4][16][PS];     // per-wave P staging (C->A relayout)

  const int tid  = threadIdx.x;
  const int lane = tid & 31;
  const int w    = tid >> 5;
  const int n15  = lane & 15;
  const int hi16 = lane >> 4;       // 0 | 1
  const int abase = hi16 * 8;       // A-fragment K interleave base (halves)
  const int bbase = hi16 * 16;      // unused: B fragments load with abase (same K interleave as A)

  const int bh    = blockIdx.y;
  const int qbase = blockIdx.x * QTILE;
  const int qw    = qbase + w * 16;               // this wave's first q row

  const size_t hoff = (size_t)bh * Sc * Dk;
  const float* Qh = Qg + hoff;
  const float* Kh = Kg + hoff;
  const float* Vh = Vg + hoff;
  float*       Oh = Og + hoff;

  // ---- Q fragments: 16x64, f32 -> f16, pre-scaled by 1/sqrt(DK) ----
  v16h aQ[2];
  {
    const float* qrow = Qh + (size_t)(qw + n15) * Dk;
    #pragma unroll
    for (int c = 0; c < 2; ++c) {
      v16h a;
      #pragma unroll
      for (int i = 0; i < 8; ++i) {
        a[i]     = (_Float16)(qrow[c * 32 + abase + i]      * 0.125f);
        a[i + 8] = (_Float16)(qrow[c * 32 + abase + 16 + i] * 0.125f);
      }
      aQ[c] = a;
    }
  }

  // all-ones B fragment: row-sum side-car matmul (P * Ones -> rowsum in every lane)
  v16h vOnes;
  #pragma unroll
  for (int i = 0; i < 16; ++i) vOnes[i] = (_Float16)1.0f;

  v8f acc[4] = {};                 // O accumulator: 16(q) x 64(dk) in 4 C-frags
  v8f accS   = {};                 // rowsum accumulator (same C layout, all columns equal)
  float rowmax[8];
  #pragma unroll
  for (int r = 0; r < 8; ++r) rowmax[r] = -INFINITY;

  const int ntiles = blockIdx.x + 1;        // keys 0 .. qbase+63 (causal upper bound)
  for (int t = 0; t < ntiles; ++t) {
    const int kbase = t * KTILE;

    __syncthreads();   // previous tile fully consumed before overwrite
    // ---- cooperative stage: K tile (row major) + V tile (transposed), f32->f16 ----
    #pragma unroll
    for (int i = 0; i < 8; ++i) {
      const int e   = (tid + i * 128) * 4;     // [0, 4096) step 4
      const int key = e >> 6;
      const int dk  = e & 63;
      const float* kp = Kh + (size_t)(kbase + key) * Dk + dk;
      const float* vp = Vh + (size_t)(kbase + key) * Dk + dk;
      const float4 kv = *(const float4*)kp;
      const float4 vv = *(const float4*)vp;
      Ksh[key][dk + 0] = (_Float16)kv.x;
      Ksh[key][dk + 1] = (_Float16)kv.y;
      Ksh[key][dk + 2] = (_Float16)kv.z;
      Ksh[key][dk + 3] = (_Float16)kv.w;
      Vt[dk + 0][key] = (_Float16)vv.x;
      Vt[dk + 1][key] = (_Float16)vv.y;
      Vt[dk + 2][key] = (_Float16)vv.z;
      Vt[dk + 3][key] = (_Float16)vv.w;
      if (kbase + KTILE < Sc) {              // warm L2/WGP$ for next tile
        __builtin_prefetch(kp + KTILE * Dk, 0, 1);
        __builtin_prefetch(vp + KTILE * Dk, 0, 1);
      }
    }
    __syncthreads();

    if (kbase > qw + 15) continue;           // tile entirely above causal diagonal (wave-uniform)

    // ---- scores: S = Q * K^T for four 16-key subtiles ----
    v8f s[4];
    #pragma unroll
    for (int st = 0; st < 4; ++st) {
      const _Float16* kr = &Ksh[st * 16 + n15][0];
      const v16h b0 = cat16(*(const h8*)(kr + 0  + abase), *(const h8*)(kr + 0  + abase + 16));
      const v16h b1 = cat16(*(const h8*)(kr + 32 + abase), *(const h8*)(kr + 32 + abase + 16));
      v8f z = {};
      z = wmma_f16(aQ[0], b0, z);
      z = wmma_f16(aQ[1], b1, z);
      s[st] = z;
    }

    // ---- causal mask: only the diagonal tile needs it (wave-uniform branch) ----
    if (kbase + KTILE - 1 > qw) {
      #pragma unroll
      for (int r = 0; r < 8; ++r) {
        const int q = qw + r + 8 * hi16;
        #pragma unroll
        for (int j = 0; j < 4; ++j)
          if (kbase + j * 16 + n15 > q) s[j][r] = -3.0e38f;
      }
    }

    // ---- online softmax: max via ds_swizzle tree; sum via ones-WMMA side-car ----
    #pragma unroll
    for (int r = 0; r < 8; ++r) {
      float mx = fmaxf(fmaxf(s[0][r], s[1][r]), fmaxf(s[2][r], s[3][r]));
      mx = fmaxf(mx, SWZ_XOR(mx, 1));
      mx = fmaxf(mx, SWZ_XOR(mx, 2));
      mx = fmaxf(mx, SWZ_XOR(mx, 4));
      mx = fmaxf(mx, SWZ_XOR(mx, 8));
      const float nm = fmaxf(rowmax[r], mx);
      const float sc = __builtin_exp2f((rowmax[r] - nm) * LOG2E);
      rowmax[r] = nm;
      const float p0 = __builtin_exp2f((s[0][r] - nm) * LOG2E);
      const float p1 = __builtin_exp2f((s[1][r] - nm) * LOG2E);
      const float p2 = __builtin_exp2f((s[2][r] - nm) * LOG2E);
      const float p3 = __builtin_exp2f((s[3][r] - nm) * LOG2E);
      accS[r]   *= sc;
      #pragma unroll
      for (int f = 0; f < 4; ++f) acc[f][r] *= sc;
      const int m = r + 8 * hi16;            // C-layout row -> LDS row
      Psh[w][m][n15]      = (_Float16)p0;
      Psh[w][m][16 + n15] = (_Float16)p1;
      Psh[w][m][32 + n15] = (_Float16)p2;
      Psh[w][m][48 + n15] = (_Float16)p3;
    }

    // ---- relayout P (C -> A fragment) via wave-private LDS ----
    const _Float16* pr = &Psh[w][n15][0];
    const v16h aP0 = cat16(*(const h8*)(pr + abase),      *(const h8*)(pr + abase + 16));
    const v16h aP1 = cat16(*(const h8*)(pr + 32 + abase), *(const h8*)(pr + 32 + abase + 16));

    // ---- O += P * V  and  rowsum += P * Ones ----
    #pragma unroll
    for (int f = 0; f < 4; ++f) {
      const _Float16* vr = &Vt[f * 16 + n15][0];
      const v16h bV0 = cat16(*(const h8*)(vr + abase),      *(const h8*)(vr + abase + 16));
      const v16h bV1 = cat16(*(const h8*)(vr + 32 + abase), *(const h8*)(vr + 32 + abase + 16));
      acc[f] = wmma_f16(aP0, bV0, acc[f]);
      acc[f] = wmma_f16(aP1, bV1, acc[f]);
    }
    accS = wmma_f16(aP0, vOnes, accS);
    accS = wmma_f16(aP1, vOnes, accS);
  }

  // ---- epilogue: normalize and store fp32 output ----
  #pragma unroll
  for (int r = 0; r < 8; ++r) {
    const float inv = 1.0f / accS[r];
    const int   q   = qw + r + 8 * hi16;
    float* orow = Oh + (size_t)q * Dk;
    #pragma unroll
    for (int f = 0; f < 4; ++f) orow[f * 16 + n15] = acc[f][r] * inv;
  }
}

extern "C" void kernel_launch(void* const* d_in, const int* in_sizes, int n_in,
                              void* d_out, int out_size, void* d_ws, size_t ws_size,
                              hipStream_t stream) {
  (void)in_sizes; (void)n_in; (void)out_size; (void)d_ws; (void)ws_size;
  // inputs: [0]=d_model, [1]=num_heads, [2]=Q, [3]=K, [4]=V, [5]=mask (causal, recomputed analytically)
  const float* Q = (const float*)d_in[2];
  const float* K = (const float*)d_in[3];
  const float* V = (const float*)d_in[4];
  float* O = (float*)d_out;
  dim3 grid(Sc / QTILE, Bc * Hc);
  fa_fwd_kernel<<<grid, dim3(128), 0, stream>>>(Q, K, V, O);
}
